// ImprovedBertSelfAttention_42185168782009
// MI455X (gfx1250) — matrix operand order repaired and checked by reading, not compiled, not run on hardware
//
#include <hip/hip_runtime.h>

// ---- problem constants -----------------------------------------------------
#define HIDDEN 768
#define NHEADS 12
#define HDIM   64
#define SEQ    2048
#define BATCH  4
#define NH_TOT (BATCH*NHEADS)   // 48
#define M_TOT  (BATCH*SEQ)      // 8192
#define N3     (3*HIDDEN)       // 2304 (Q|K|V concatenated)

// Async global->LDS (gfx1250, ASYNCcnt) if the toolchain exposes it.
#if defined(__HIP_DEVICE_COMPILE__) && \
    __has_builtin(__builtin_amdgcn_global_load_async_to_lds_b128) && \
    __has_builtin(__builtin_amdgcn_s_wait_asynccnt)
#define USE_ASYNC 1
#else
#define USE_ASYNC 0
#endif

typedef __attribute__((ext_vector_type(16))) __bf16 v16bf;
typedef __attribute__((ext_vector_type(8)))  float  v8f;

union ABOp { v16bf v; unsigned int u[8]; };

#if USE_ASYNC
typedef int v4i_ __attribute__((vector_size(16)));
// 16B async copy global->LDS, tracked by ASYNCcnt.
__device__ __forceinline__ void async_cp16(const void* g, void* l) {
  __builtin_amdgcn_global_load_async_to_lds_b128(
      (__attribute__((address_space(1))) v4i_*)
          (__attribute__((address_space(1))) void*)(void*)g,
      (__attribute__((address_space(3))) v4i_*)
          (__attribute__((address_space(3))) void*)(void*)l,
      0, 0);
}
#endif

__device__ __forceinline__ unsigned short f32_bf16(float f) {
  unsigned int x = __float_as_uint(f);
  x += 0x7FFFu + ((x >> 16) & 1u);          // round-to-nearest-even
  return (unsigned short)(x >> 16);
}

// A-operand (16x32 bf16, wave32): lane L -> row M=L&15; VGPR v holds K pair:
// K = 2*(v&3) + 16*(v>>2) + 8*(L>>4)   (per CDNA5 ISA 7.12.2 16-bit A table)
__device__ __forceinline__ int kmapA(int v, int h) {
  return 2*(v & 3) + ((v >> 2) << 4) + (h << 3);
}

__device__ __forceinline__ v8f wmma_bf16(const ABOp& a, const ABOp& b, v8f c) {
  return __builtin_amdgcn_wmma_f32_16x16x32_bf16(
      /*neg_a=*/false, a.v, /*neg_b=*/false, b.v,
      /*c_mod=*/(short)0, c, /*reuse_a=*/false, /*reuse_b=*/false);
}

__device__ __forceinline__ float half16_max(float v) {
#pragma unroll
  for (int m = 1; m <= 8; m <<= 1) v = fmaxf(v, __shfl_xor(v, m, 32));
  return v;
}
__device__ __forceinline__ float half16_sum(float v) {
#pragma unroll
  for (int m = 1; m <= 8; m <<= 1) v += __shfl_xor(v, m, 32);
  return v;
}

// ---- kernel 1: fp32 -> bf16 conversion of hidden states --------------------
__global__ void cvt_x_kernel(const float* __restrict__ x,
                             unsigned short* __restrict__ xb, int n) {
  int i = blockIdx.x * blockDim.x + threadIdx.x;
  if (i < n) xb[i] = f32_bf16(x[i]);
}

// ---- kernel 2: build transposed concatenated weight WT[k][n], n in [0,2304)
__global__ void build_wt_kernel(const float* __restrict__ Wq,
                                const float* __restrict__ Wk,
                                const float* __restrict__ Wv,
                                unsigned short* __restrict__ wt) {
  int i = blockIdx.x * blockDim.x + threadIdx.x;   // over 768*2304
  if (i >= HIDDEN * N3) return;
  int k = i / N3, n = i - k * N3;
  int sec = n / HIDDEN, nc = n - sec * HIDDEN;
  const float* W = (sec == 0) ? Wq : (sec == 1) ? Wk : Wv;
  wt[i] = f32_bf16(W[nc * HIDDEN + k]);
}

// ---- tile staging helpers (async global->LDS when available) ---------------
// GEMM: WT chunk [32 k][64 n] bf16 = 4KB; one b128 per thread.
__device__ __forceinline__ void stage_b(const unsigned short* __restrict__ wt,
                                        int kc, int nbase,
                                        unsigned short* bb, int tid) {
#if USE_ASYNC
  int row = tid >> 3, c = (tid & 7) << 3;          // 32 rows x 8 chunks(8 shorts)
  async_cp16(wt + (kc + row) * N3 + nbase + c, bb + (tid << 3));
#else
  unsigned int* dst = (unsigned int*)bb;
#pragma unroll
  for (int i = 0; i < 4; ++i) {
    int idx = i * 256 + tid;
    int row = idx >> 5, col2 = idx & 31;
    dst[idx] = *(const unsigned int*)&wt[(kc + row) * N3 + nbase + col2 * 2];
  }
#endif
}

// Flash: K^T chunk [64 hd][32 key] + V chunk [32 key][64 hd]; 2 b128/thread.
__device__ __forceinline__ void stage_kv(const unsigned short* __restrict__ kt,
                                         const unsigned short* __restrict__ vh,
                                         int kb, unsigned short* bk,
                                         unsigned short* bv, int tid) {
#if USE_ASYNC
  int rK = tid >> 2, cK = (tid & 3) << 3;          // 64 rows x 4 chunks
  async_cp16(kt + rK * SEQ + kb + cK, bk + (tid << 3));
  int rV = tid >> 3, cV = (tid & 7) << 3;          // 32 rows x 8 chunks
  async_cp16(vh + (kb + rV) * HDIM + cV, bv + (tid << 3));
#else
  unsigned int* dK = (unsigned int*)bk;
  unsigned int* dV = (unsigned int*)bv;
#pragma unroll
  for (int i = 0; i < 4; ++i) {
    int idx = i * 256 + tid;
    int rK = idx >> 4, cK = idx & 15;
    dK[idx] = *(const unsigned int*)&kt[rK * SEQ + kb + cK * 2];
    int rV = idx >> 5, cV = idx & 31;
    dV[idx] = *(const unsigned int*)&vh[(kb + rV) * HDIM + cV * 2];
  }
#endif
}

// ---- kernel 3: fused QKV projection GEMM (double-buffered B staging) -------
__global__ __launch_bounds__(256)
void qkv_gemm_kernel(const unsigned short* __restrict__ xb,
                     const unsigned short* __restrict__ wt,
                     const float* __restrict__ bq, const float* __restrict__ bk,
                     const float* __restrict__ bv,
                     unsigned short* __restrict__ qws,
                     unsigned short* __restrict__ kws,
                     unsigned short* __restrict__ vws) {
  __shared__ __align__(16) unsigned short ldsB[2][32 * 64];
  const int tid = threadIdx.x;
  const int wave = tid >> 5, lane = tid & 31;
  const int half = lane >> 4, l16 = lane & 15;
  const int rbase = blockIdx.x * 128 + wave * 16;
  const int nbase = blockIdx.y * 64;

  v8f acc[4] = {};

  stage_b(wt, 0, nbase, ldsB[0], tid);
  const int NKC = HIDDEN / 32;                      // 24 k-chunks
  for (int it = 0; it < NKC; ++it) {
    const int kc = it * 32;
    const int cur = it & 1;
    if (it + 1 < NKC) {
      stage_b(wt, kc + 32, nbase, ldsB[1 - cur], tid);
#if USE_ASYNC
      __builtin_amdgcn_s_wait_asynccnt(1);          // current chunk complete
#endif
    } else {
#if USE_ASYNC
      __builtin_amdgcn_s_wait_asynccnt(0);
#endif
    }
    __syncthreads();

    ABOp a;                                         // X rows rbase..+15
    {
      const unsigned short* xrow = xb + (rbase + l16) * HIDDEN + kc;
#pragma unroll
      for (int v = 0; v < 8; ++v)
        a.u[v] = *(const unsigned int*)&xrow[kmapA(v, half)];
    }
#pragma unroll
    for (int t = 0; t < 4; ++t) {
      ABOp b;                                       // B: lane = n, vgpr = k pair (kmapA)
#pragma unroll
      for (int v = 0; v < 8; ++v) {
        int kk = kmapA(v, half);
        b.u[v] = (unsigned int)ldsB[cur][kk * 64 + t * 16 + l16] |
                 ((unsigned int)ldsB[cur][(kk + 1) * 64 + t * 16 + l16] << 16);
      }
      acc[t] = wmma_bf16(a, b, acc[t]);
    }
    __syncthreads();
  }

#pragma unroll
  for (int t = 0; t < 4; ++t) {
    int nglob = nbase + t * 16 + l16;
    int sec = nglob / HIDDEN;
    int ncol = nglob - sec * HIDDEN;
    int h = ncol >> 6, d = ncol & 63;
    float bias = ((sec == 0) ? bq : (sec == 1) ? bk : bv)[ncol];
#pragma unroll
    for (int r = 0; r < 8; ++r) {
      int m = rbase + r + 8 * half;                 // C layout: row = r + 8*half
      int b = m >> 11, s = m & 2047;
      unsigned short us = f32_bf16(acc[t][r] + bias);
      int head = b * NHEADS + h;
      if (sec == 0)      qws[((head * SEQ + s) << 6) + d] = us;
      else if (sec == 1) kws[(head * HDIM + d) * SEQ + s] = us;
      else               vws[((head * SEQ + s) << 6) + d] = us;
    }
  }
}

// ---- kernel 4: flash attention (double-buffered K/V async staging) ---------
__global__ __launch_bounds__(256)
void flash_attn_kernel(const unsigned short* __restrict__ qws,
                       const unsigned short* __restrict__ kws,
                       const unsigned short* __restrict__ vws,
                       const float* __restrict__ mask,
                       const float* __restrict__ head_bias,
                       float* __restrict__ out) {
  __shared__ __align__(16) unsigned short ldsKT[2][64 * 32]; // [hd][key']
  __shared__ __align__(16) unsigned short ldsV [2][32 * 64]; // [key'][hd]
  __shared__ __align__(16) unsigned short ldsP [8][16 * 32]; // per-wave P tile

  const int tid = threadIdx.x, wave = tid >> 5, lane = tid & 31;
  const int half = lane >> 4, l16 = lane & 15;
  const int head = blockIdx.x;                      // 0..47
  const int b = head / NHEADS, h = head - b * NHEADS;
  const int qbase = blockIdx.y * 128 + wave * 16;

  const unsigned short* qh = qws + (size_t)head * SEQ * HDIM;
  const unsigned short* kt = kws + (size_t)head * HDIM * SEQ;
  const unsigned short* vh = vws + (size_t)head * SEQ * HDIM;
  const float* mrow = mask + b * SEQ;
  const float hb = 0.5f * head_bias[h];
  const float scl = 0.125f;                         // 1/sqrt(64)

  // Q A-operands (16 rows x 64 hd = two 16x32 chunks), loaded once.
  ABOp qa0, qa1;
  {
    const unsigned short* qrow = qh + (qbase + l16) * HDIM;
#pragma unroll
    for (int v = 0; v < 8; ++v) {
      int kk = kmapA(v, half);
      qa0.u[v] = *(const unsigned int*)&qrow[kk];
      qa1.u[v] = *(const unsigned int*)&qrow[32 + kk];
    }
  }

  v8f o[4] = {};
  float mrun[8], lrun[8];
#pragma unroll
  for (int r = 0; r < 8; ++r) { mrun[r] = -3.0e38f; lrun[r] = 0.f; }

  stage_kv(kt, vh, 0, ldsKT[0], ldsV[0], tid);
  const int NIT = SEQ / 32;                         // 64 key chunks
  for (int it = 0; it < NIT; ++it) {
    const int kb = it * 32;
    const int cur = it & 1;
    if (it + 1 < NIT) {
      stage_kv(kt, vh, kb + 32, ldsKT[1 - cur], ldsV[1 - cur], tid);
#if USE_ASYNC
      __builtin_amdgcn_s_wait_asynccnt(2);          // current chunk complete
#endif
    } else {
#if USE_ASYNC
      __builtin_amdgcn_s_wait_asynccnt(0);
#endif
    }
    __syncthreads();

    // scores: two 16x16 tiles (keys kb..+15, kb+16..+31), hd split 2x32
    v8f s0 = {}, s1 = {};
#pragma unroll
    for (int c = 0; c < 2; ++c) {
      ABOp b0, b1;
#pragma unroll
      for (int v = 0; v < 8; ++v) {
        int rowk = (c * 32 + kmapA(v, half)) * 32;  // B: lane = key, vgpr = hd pair (kmapA)
        b0.u[v] = (unsigned int)ldsKT[cur][rowk + l16] |
                  ((unsigned int)ldsKT[cur][rowk + 32 + l16] << 16);
        b1.u[v] = (unsigned int)ldsKT[cur][rowk + 16 + l16] |
                  ((unsigned int)ldsKT[cur][rowk + 48 + l16] << 16);
      }
      ABOp aq = c ? qa1 : qa0;
      s0 = wmma_bf16(aq, b0, s0);
      s1 = wmma_bf16(aq, b1, s1);
    }

    float add0 = (1.f - mrow[kb + l16]) * -10000.f + hb;
    float add1 = (1.f - mrow[kb + 16 + l16]) * -10000.f + hb;

#pragma unroll
    for (int r = 0; r < 8; ++r) {
      float v0 = s0[r] * scl + add0;
      float v1 = s1[r] * scl + add1;
      float mnew = fmaxf(mrun[r], half16_max(fmaxf(v0, v1)));
      float alpha = __expf(mrun[r] - mnew);
      float p0 = __expf(v0 - mnew);
      float p1 = __expf(v1 - mnew);
      lrun[r] = alpha * lrun[r] + half16_sum(p0 + p1);
      mrun[r] = mnew;
#pragma unroll
      for (int j = 0; j < 4; ++j) o[j][r] *= alpha;
      int row = (r + 8 * half) * 32;                // C layout -> LDS row-major
      ldsP[wave][row + l16]      = f32_bf16(p0);
      ldsP[wave][row + 16 + l16] = f32_bf16(p1);
    }

    ABOp pa;                                        // re-read P in A layout
#pragma unroll
    for (int v = 0; v < 8; ++v)
      pa.u[v] = *(const unsigned int*)&ldsP[wave][l16 * 32 + kmapA(v, half)];

#pragma unroll
    for (int j = 0; j < 4; ++j) {
      ABOp bv_;                                     // B: lane = hd, vgpr = key' pair (kmapA)
#pragma unroll
      for (int v = 0; v < 8; ++v) {
        int kk = kmapA(v, half);
        bv_.u[v] = (unsigned int)ldsV[cur][kk * 64 + j * 16 + l16] |
                   ((unsigned int)ldsV[cur][(kk + 1) * 64 + j * 16 + l16] << 16);
      }
      o[j] = wmma_bf16(pa, bv_, o[j]);
    }
    __syncthreads();
  }

#pragma unroll
  for (int j = 0; j < 4; ++j) {
    int col = h * HDIM + j * 16 + l16;
#pragma unroll
    for (int r = 0; r < 8; ++r) {
      int row = qbase + r + 8 * half;
      out[(size_t)(b * SEQ + row) * HIDDEN + col] = o[j][r] / lrun[r];
    }
  }
}

// ---- launcher --------------------------------------------------------------
extern "C" void kernel_launch(void* const* d_in, const int* in_sizes, int n_in,
                              void* d_out, int out_size, void* d_ws, size_t ws_size,
                              hipStream_t stream) {
  const float* hs    = (const float*)d_in[0];
  const float* mask  = (const float*)d_in[1];
  const float* Wq    = (const float*)d_in[2];
  const float* bq    = (const float*)d_in[3];
  const float* Wk    = (const float*)d_in[4];
  const float* bk    = (const float*)d_in[5];
  const float* Wv    = (const float*)d_in[6];
  const float* bv    = (const float*)d_in[7];
  const float* hbias = (const float*)d_in[8];
  float* out = (float*)d_out;

  // workspace layout (all sizes multiples of 256B):
  char* ws = (char*)d_ws;
  unsigned short* xb  = (unsigned short*)(ws);                        // 12.6 MB
  unsigned short* wt  = (unsigned short*)(ws + 12582912);             //  3.5 MB
  unsigned short* qws = (unsigned short*)(ws + 12582912 + 3538944);   // 12.6 MB
  unsigned short* kws = qws + (size_t)NH_TOT * SEQ * HDIM;            // 12.6 MB
  unsigned short* vws = kws + (size_t)NH_TOT * SEQ * HDIM;            // 12.6 MB

  cvt_x_kernel<<<(M_TOT * HIDDEN + 255) / 256, 256, 0, stream>>>(
      hs, xb, M_TOT * HIDDEN);
  build_wt_kernel<<<(HIDDEN * N3 + 255) / 256, 256, 0, stream>>>(
      Wq, Wk, Wv, wt);
  qkv_gemm_kernel<<<dim3(M_TOT / 128, N3 / 64), 256, 0, stream>>>(
      xb, wt, bq, bk, bv, qws, kws, vws);
  flash_attn_kernel<<<dim3(NH_TOT, SEQ / 128), 256, 0, stream>>>(
      qws, kws, vws, mask, hbias, out);
}
